// Attention_21981642621177
// MI455X (gfx1250) — matrix operand order repaired and checked by reading, not compiled, not run on hardware
//
#include <hip/hip_runtime.h>

// LeViT attention, fully fused: one workgroup (256 thr = 8 wave32) per image.
// All intermediates live in LDS (304KB of the WGP's 320KB); math via
// v_wmma_f32_16x16x32_f16 with fp32 accumulate + fp32 softmax.

#define DIMC  384
#define NTOK  49
#define HEADS 8
#define KD    32
#define DV    128
#define DHC   1024
#define NQK   256
#define SCALEF 0.17677669529663687f  // 32^-0.5

typedef __attribute__((ext_vector_type(16))) _Float16 v16h;
typedef __attribute__((ext_vector_type(8)))  float    v8f;

#define WMMA(a, b, c) \
  __builtin_amdgcn_wmma_f32_16x16x32_f16(false, (a), false, (b), (short)0, (c), false, false)

// ---- fragment loaders ---------------------------------------------------
// A (16x32 f16, MxK): lane holds row m=lane&15; elems 0..15 -> K = half*16+e.
// `row` = &A[m][k0] for this lane.
static __device__ __forceinline__ v16h load_a16(const _Float16* row, int half) {
  const int k0 = half * 16, k1 = half * 16 + 8;
  v16h a;
#pragma unroll
  for (int e = 0; e < 8; ++e) { a[e] = row[k0 + e]; a[8 + e] = row[k1 + e]; }
  return a;
}
// B (32x16 f16, KxN) read from Bt[N][K] row-major: lane holds col n=lane&15,
// K = half*16 + e (16 contiguous f16).  `row` = &Bt[n][k0] for this lane.
static __device__ __forceinline__ v16h load_b16(const _Float16* row, int half) {
  const _Float16* r = row + half * 16;
  v16h b;
#pragma unroll
  for (int e = 0; e < 16; ++e) b[e] = r[e];
  return b;
}
// Same patterns, but sourced from fp32 global weights with on-the-fly f16 cvt.
static __device__ __forceinline__ v16h load_a32(const float* row, int half) {
  const int k0 = half * 16, k1 = half * 16 + 8;
  v16h a;
#pragma unroll
  for (int e = 0; e < 8; ++e) {
    a[e]     = (_Float16)row[k0 + e];
    a[8 + e] = (_Float16)row[k1 + e];
  }
  return a;
}
static __device__ __forceinline__ v16h load_b32(const float* row, int half) {
  const float* r = row + half * 16;
  v16h b;
#pragma unroll
  for (int e = 0; e < 16; ++e) b[e] = (_Float16)r[e];
  return b;
}

__global__ __launch_bounds__(256, 1) void levit_attn_kernel(
    const float* __restrict__ x,
    const float* __restrict__ q_w, const float* __restrict__ q_b,
    const float* __restrict__ k_w, const float* __restrict__ k_b,
    const float* __restrict__ v_w, const float* __restrict__ v_b,
    const float* __restrict__ proj_w, const float* __restrict__ proj_b,
    const float* __restrict__ abias, const int* __restrict__ bidx,
    float* __restrict__ out, int n_off) {
  // LDS layout (bytes):
  //   [0      , 49152)  xs  : x   f16 [64][384]
  //   [49152  , 81920)  qs  : q   f16 [64][256]
  //   [81920  ,114688)  ks  : k   f16 [64][256]
  //   [114688 ,180224)  Ps  : P   f16 [8][64][64]
  //   [180224 ,311296)  vTs : v^T f16 [8][128][64]
  //   os (O staging, f16 [64][1024]) overlays offset 0; only rows<49 are
  //   written (max byte 100350 < 114688) so live Ps/vTs are never clobbered.
  __shared__ __align__(16) unsigned char smem[311296];
  _Float16* xs  = (_Float16*)smem;
  _Float16* qs  = (_Float16*)(smem + 49152);
  _Float16* ks  = (_Float16*)(smem + 81920);
  _Float16* Ps  = (_Float16*)(smem + 114688);
  _Float16* vTs = (_Float16*)(smem + 180224);
  _Float16* os  = (_Float16*)smem;

  const int tid  = threadIdx.x;
  const int wave = tid >> 5;   // == head index
  const int lane = tid & 31;
  const int half = lane >> 4;
  const int ln   = lane & 15;
  const int b    = blockIdx.x;

  // ---------------- load x tile (f32 -> f16), zero the 15 pad rows --------
  const float* xb = x + (size_t)b * NTOK * DIMC;
  for (int i = tid; i < NTOK * DIMC; i += 256) xs[i] = (_Float16)xb[i];
  for (int i = tid; i < (64 - NTOK) * DIMC; i += 256)
    xs[NTOK * DIMC + i] = (_Float16)0.f;
  __syncthreads();

  // ---------------- phase 1a: [Q|K] = x @ [q_w|k_w]^T + bias --------------
  // 512 output cols = 32 N-tiles; 4 per wave.  Bt = weight rows (natural).
  for (int t = wave * 4; t < wave * 4 + 4; ++t) {
    const int  n0  = t * 16;
    const bool isq = (n0 < NQK);
    const float* W  = isq ? (q_w + (size_t)n0 * DIMC)
                          : (k_w + (size_t)(n0 - NQK) * DIMC);
    const float  bv = isq ? q_b[n0 + ln] : k_b[n0 - NQK + ln];
    v8f acc[4];
#pragma unroll
    for (int mt = 0; mt < 4; ++mt)
#pragma unroll
      for (int r = 0; r < 8; ++r) acc[mt][r] = bv;
    for (int kk = 0; kk < DIMC / 32; ++kk) {
      v16h bf = load_b32(W + (size_t)ln * DIMC + kk * 32, half);
#pragma unroll
      for (int mt = 0; mt < 4; ++mt) {
        v16h af = load_a16(xs + (mt * 16 + ln) * DIMC + kk * 32, half);
        acc[mt] = WMMA(af, bf, acc[mt]);
      }
    }
    _Float16* dst = isq ? (qs + n0) : (ks + (n0 - NQK));
#pragma unroll
    for (int mt = 0; mt < 4; ++mt)
#pragma unroll
      for (int r = 0; r < 8; ++r)
        dst[(mt * 16 + r + half * 8) * NQK + ln] = (_Float16)acc[mt][r];
  }

  // ---------------- phase 1b: vT_h = v_w_h @ x^T + bias (per head) --------
  // Produces V already transposed ([d][m]) so attn@V needs no transpose.
  {
    const float* Wv  = v_w + (size_t)wave * DV * DIMC;
    _Float16*    vTh = vTs + wave * DV * 64;
    for (int mt = 0; mt < DV / 16; ++mt) {  // d tiles
      v8f acc[4];
#pragma unroll
      for (int nt = 0; nt < 4; ++nt)
#pragma unroll
        for (int r = 0; r < 8; ++r)
          acc[nt][r] = v_b[wave * DV + mt * 16 + r + half * 8];
      for (int kk = 0; kk < DIMC / 32; ++kk) {
        v16h af = load_a32(Wv + (size_t)(mt * 16 + ln) * DIMC + kk * 32, half);
#pragma unroll
        for (int nt = 0; nt < 4; ++nt) {  // m tiles (Bt = x rows, natural)
          v16h bf = load_b16(xs + (nt * 16 + ln) * DIMC + kk * 32, half);
          acc[nt] = WMMA(af, bf, acc[nt]);
        }
      }
#pragma unroll
      for (int nt = 0; nt < 4; ++nt)
#pragma unroll
        for (int r = 0; r < 8; ++r)
          vTh[(mt * 16 + r + half * 8) * 64 + nt * 16 + ln] = (_Float16)acc[nt][r];
    }
  }
  __syncthreads();  // q,k,vT visible; also last cross-wave read of xs below is done

  // ---------------- phase 2: S = q_h @ k_h^T ; fp32 softmax -> P ----------
  {
    const _Float16* qh = qs + wave * KD;
    const _Float16* kh = ks + wave * KD;
    _Float16*       Ph = Ps + wave * 64 * 64;
    v16h bfs[4];
#pragma unroll
    for (int nt = 0; nt < 4; ++nt)
      bfs[nt] = load_b16(kh + (nt * 16 + ln) * NQK, half);  // Bt = k rows
    v8f s[4][4];
#pragma unroll
    for (int mt = 0; mt < 4; ++mt) {
      v16h af = load_a16(qh + (mt * 16 + ln) * NQK, half);  // K = 32 -> 1 step
#pragma unroll
      for (int nt = 0; nt < 4; ++nt) {
        v8f z;
#pragma unroll
        for (int r = 0; r < 8; ++r) z[r] = 0.f;
        s[mt][nt] = WMMA(af, bfs[nt], z);
      }
    }
    // scale + relative-position bias + mask, then row softmax.
    // C-fragment: row = mt*16 + r + half*8 (uniform over the 16-lane half),
    // col = nt*16 + ln  -> row reduce = 4-way VALU + shfl_xor(width 16).
#pragma unroll
    for (int mt = 0; mt < 4; ++mt) {
      float l[4][8];
#pragma unroll
      for (int nt = 0; nt < 4; ++nt)
#pragma unroll
        for (int r = 0; r < 8; ++r) {
          int row = mt * 16 + r + half * 8;
          int col = nt * 16 + ln;
          float v;
          if (row < NTOK && col < NTOK)
            v = s[mt][nt][r] * SCALEF +
                abias[wave * n_off + bidx[row * NTOK + col]];
          else
            v = -1e30f;  // masked: exp -> exact +0, so K=64 padding is inert
          l[nt][r] = v;
        }
#pragma unroll
      for (int r = 0; r < 8; ++r) {
        float m = fmaxf(fmaxf(l[0][r], l[1][r]), fmaxf(l[2][r], l[3][r]));
#pragma unroll
        for (int off = 1; off < 16; off <<= 1) m = fmaxf(m, __shfl_xor(m, off, 16));
        float p0 = __expf(l[0][r] - m), p1 = __expf(l[1][r] - m);
        float p2 = __expf(l[2][r] - m), p3 = __expf(l[3][r] - m);
        float sum = p0 + p1 + p2 + p3;
#pragma unroll
        for (int off = 1; off < 16; off <<= 1) sum += __shfl_xor(sum, off, 16);
        float inv = 1.f / sum;
        int row = mt * 16 + r + half * 8;
        Ph[row * 64 +  0 + ln] = (_Float16)(p0 * inv);
        Ph[row * 64 + 16 + ln] = (_Float16)(p1 * inv);
        Ph[row * 64 + 32 + ln] = (_Float16)(p2 * inv);
        Ph[row * 64 + 48 + ln] = (_Float16)(p3 * inv);
      }
    }
  }
  __syncthreads();  // all waves done reading qs/ks before os overlays them

  // ---------------- phase 3: O_h = P_h @ V_h  (A=P, Bt=vT, both natural) --
  {
    const _Float16* Ph  = Ps + wave * 64 * 64;
    const _Float16* vTh = vTs + wave * DV * 64;
    for (int c = 0; c < 2; ++c) {  // 2 chunks of 64 d-cols to cap VGPR use
      v8f o[4][4];
#pragma unroll
      for (int mt = 0; mt < 4; ++mt)
#pragma unroll
        for (int nt = 0; nt < 4; ++nt)
#pragma unroll
          for (int r = 0; r < 8; ++r) o[mt][nt][r] = 0.f;
#pragma unroll
      for (int kt = 0; kt < 2; ++kt) {  // K = 64 (padded m)
        v16h af[4];
#pragma unroll
        for (int mt = 0; mt < 4; ++mt)
          af[mt] = load_a16(Ph + (mt * 16 + ln) * 64 + kt * 32, half);
#pragma unroll
        for (int nt = 0; nt < 4; ++nt) {
          v16h bf = load_b16(vTh + (c * 64 + nt * 16 + ln) * 64 + kt * 32, half);
#pragma unroll
          for (int mt = 0; mt < 4; ++mt) o[mt][nt] = WMMA(af[mt], bf, o[mt][nt]);
        }
      }
#pragma unroll
      for (int mt = 0; mt < 4; ++mt)
#pragma unroll
        for (int nt = 0; nt < 4; ++nt)
#pragma unroll
          for (int r = 0; r < 8; ++r) {
            int row = mt * 16 + r + half * 8;
            if (row < NTOK)  // rows>=49 skipped => never touches live Ps/vTs
              os[row * DHC + wave * DV + c * 64 + nt * 16 + ln] =
                  (_Float16)o[mt][nt][r];
          }
    }
  }
  __syncthreads();  // full O[49][1024] staged

  // ---------------- phase 4: out = O @ proj_w^T + proj_b ------------------
  // 384 cols = 24 N-tiles; 3 per wave.  Bt = proj_w rows (natural).
  float* ob = out + (size_t)b * NTOK * DIMC;
  for (int t = wave * 3; t < wave * 3 + 3; ++t) {
    const int n0 = t * 16;
    const float pb = proj_b[n0 + ln];
    v8f acc[4];
#pragma unroll
    for (int mt = 0; mt < 4; ++mt)
#pragma unroll
      for (int r = 0; r < 8; ++r) acc[mt][r] = pb;
    for (int kk = 0; kk < DHC / 32; ++kk) {
      v16h bf = load_b32(proj_w + (size_t)(n0 + ln) * DHC + kk * 32, half);
#pragma unroll
      for (int mt = 0; mt < 4; ++mt) {
        v16h af = load_a16(os + (mt * 16 + ln) * DHC + kk * 32, half);
        acc[mt] = WMMA(af, bf, acc[mt]);
      }
    }
#pragma unroll
    for (int mt = 0; mt < 4; ++mt)
#pragma unroll
      for (int r = 0; r < 8; ++r) {
        int row = mt * 16 + r + half * 8;
        if (row < NTOK) ob[(size_t)row * DIMC + n0 + ln] = acc[mt][r];
      }
  }
}

extern "C" void kernel_launch(void* const* d_in, const int* in_sizes, int n_in,
                              void* d_out, int out_size, void* d_ws, size_t ws_size,
                              hipStream_t stream) {
  const float* x      = (const float*)d_in[0];
  const float* q_w    = (const float*)d_in[1];
  const float* q_b    = (const float*)d_in[2];
  const float* k_w    = (const float*)d_in[3];
  const float* k_b    = (const float*)d_in[4];
  const float* v_w    = (const float*)d_in[5];
  const float* v_b    = (const float*)d_in[6];
  const float* proj_w = (const float*)d_in[7];
  const float* proj_b = (const float*)d_in[8];
  const float* abias  = (const float*)d_in[9];
  const int*   bidx   = (const int*)d_in[10];
  float*       out    = (float*)d_out;

  const int B     = in_sizes[0] / (NTOK * DIMC);
  const int n_off = in_sizes[9] / HEADS;

  levit_attn_kernel<<<B, 256, 0, stream>>>(x, q_w, q_b, k_w, k_b, v_w, v_b,
                                           proj_w, proj_b, abias, bidx, out,
                                           n_off);
}
